// ResGraphModule_32701880991772
// MI455X (gfx1250) — matrix operand order repaired and checked by reading, not compiled, not run on hardware
//
#include <hip/hip_runtime.h>

// Problem constants (match reference)
#define N_NODES 50000
#define N_EDGES 800000
#define DIN     128
#define DOUT    16
#define NH      8
#define DED     32
#define HO      128   // NH * DOUT

typedef __attribute__((ext_vector_type(16))) __bf16 v16bf;
typedef __attribute__((ext_vector_type(8)))  __bf16 v8bf;
typedef __attribute__((ext_vector_type(2)))  __bf16 v2bf;
typedef __attribute__((ext_vector_type(8)))  float  v8f;

// -------- order-preserving float<->uint key for atomic segment-max --------
__device__ __forceinline__ unsigned fkey(float f) {
  unsigned u = __float_as_uint(f);
  return (u & 0x80000000u) ? ~u : (u | 0x80000000u);
}
__device__ __forceinline__ float fkey_dec(unsigned k) {
  return (k & 0x80000000u) ? __uint_as_float(k & 0x7FFFFFFFu)
                           : __uint_as_float(~k);
}

// Build a 16x32 A-fragment row from two contiguous 8-bf16 runs.
__device__ __forceinline__ v16bf make_a_frag(const __bf16* __restrict__ row) {
  v8bf lo = *(const v8bf*)(row);        // K offsets +0..7
  v8bf hi = *(const v8bf*)(row + 16);   // K offsets +16..23
  v16bf a;
#pragma unroll
  for (int i = 0; i < 8; ++i) { a[i] = lo[i]; a[i + 8] = hi[i]; }
  return a;
}

// ---------------- Prep 1: fp32 -> bf16 streaming convert (pairs) ----------
__global__ __launch_bounds__(256)
void cvt_bf16_kernel(const float* __restrict__ in, __bf16* __restrict__ out,
                     long long npairs) {
  long long i = (long long)blockIdx.x * blockDim.x + threadIdx.x;
  if (i >= npairs) return;
  float2 t = ((const float2*)in)[i];
  v2bf o;
  o[0] = (__bf16)t.x;
  o[1] = (__bf16)t.y;
  ((v2bf*)out)[i] = o;
}

// ---------------- Prep 2: pack weight matrix into B-fragment order --------
// out layout: [ct * (K/32) + kci][lane][16 bf16]  (32 bytes per lane, contiguous)
// B lane mapping: col = ct*16 + (lane&15); VGPR v holds K = kci*32 + (lane>>4)*8 + 2v + (v>=4 ? 8 : 0), +1
__global__ __launch_bounds__(256)
void pack_w_kernel(const float* __restrict__ W, __bf16* __restrict__ out,
                   int K, int CT, int ldw) {
  int t = blockIdx.x * blockDim.x + threadIdx.x;
  int nt = CT * (K / 32) * 32;
  if (t >= nt) return;
  int lane = t & 31;
  int tile = t >> 5;
  int kcn  = K / 32;
  int ct   = tile / kcn;
  int kci  = tile - ct * kcn;
  int half = lane >> 4;
  int lidx = lane & 15;
  int col  = ct * 16 + lidx;
  int kb   = kci * 32 + half * 8;
  __bf16* o = out + (size_t)t * 16;
#pragma unroll
  for (int v = 0; v < 8; ++v) {
    o[2 * v]     = (__bf16)W[(size_t)(kb + 2 * v + (v >= 4 ? 8 : 0)) * ldw + col];
    o[2 * v + 1] = (__bf16)W[(size_t)(kb + 2 * v + 1 + (v >= 4 ? 8 : 0)) * ldw + col];
  }
}

// ---------------- Kernel 1: node projections via WMMA bf16 ----------------
// grid.x = N/16 row tiles, grid.y = 26 output tiles:
//   0..7 -> q col tiles, 8..15 -> k, 16..23 -> v, 24 -> x@Wskip+bskip, 25 -> x@Wres
__global__ __launch_bounds__(32)
void node_proj_kernel(const __bf16* __restrict__ xbf,
                      const __bf16* __restrict__ wqp,
                      const __bf16* __restrict__ wkp,
                      const __bf16* __restrict__ wvp,
                      const __bf16* __restrict__ wsp,
                      const __bf16* __restrict__ wrp,
                      const float* __restrict__ bq, const float* __restrict__ bk,
                      const float* __restrict__ bv, const float* __restrict__ bskip,
                      float* __restrict__ qb, float* __restrict__ kb,
                      float* __restrict__ vb,
                      float* __restrict__ s1, float* __restrict__ s2) {
  const int lane = threadIdx.x & 31;
  const int half = lane >> 4;
  const int lidx = lane & 15;
  const int row0 = blockIdx.x * 16;
  const int z    = blockIdx.y;

  const __bf16* wpk; const float* bias; float* Y; int ct;
  if (z < 8)       { wpk = wqp; bias = bq;      Y = qb; ct = z;      }
  else if (z < 16) { wpk = wkp; bias = bk;      Y = kb; ct = z - 8;  }
  else if (z < 24) { wpk = wvp; bias = bv;      Y = vb; ct = z - 16; }
  else if (z == 24){ wpk = wsp; bias = bskip;   Y = s1; ct = 0;      }
  else             { wpk = wrp; bias = nullptr; Y = s2; ct = 0;      }

  const int col = ct * 16 + lidx;
  v8f acc = {};

#pragma unroll
  for (int kci = 0; kci < 4; ++kci) {   // K = 128 = 4 x 32
    v16bf a = make_a_frag(xbf + (size_t)(row0 + lidx) * DIN + kci * 32 + half * 8);
    v16bf b = *(const v16bf*)(wpk + ((size_t)(ct * 4 + kci) * 32 + lane) * 16);
    acc = __builtin_amdgcn_wmma_f32_16x16x32_bf16(false, a, false, b,
                                                  (short)0, acc, false, false);
  }

  const float bv_ = bias ? bias[col] : 0.0f;
  // Constant-stride stores (immediate offsets, no 64-bit add chains)
  if (z < 24) {
    float* yp = Y + (size_t)row0 * HO + half * (8 * HO) + col;
#pragma unroll
    for (int v = 0; v < 8; ++v) yp[v * HO] = acc[v] + bv_;
  } else {
    float* yp = Y + (size_t)row0 * DOUT + half * (8 * DOUT) + col;
#pragma unroll
    for (int v = 0; v < 8; ++v) yp[v * DOUT] = acc[v] + bv_;
  }
}

// ---------------- Kernel 2: zero/init accumulators ----------------
__global__ __launch_bounds__(256)
void init_kernel(unsigned* __restrict__ smaxkey, float* __restrict__ den,
                 float* __restrict__ agg) {
  long long i = (long long)blockIdx.x * blockDim.x + threadIdx.x;
  if (i < (long long)N_NODES * NH) { smaxkey[i] = 0u; den[i] = 0.0f; }
  if (i < (long long)N_NODES * HO) { agg[i] = 0.0f; }
}

// ---------------- Kernel 3: edge pass 1 — e=attr@We (WMMA), scores, seg-max ----
// one wave handles 16 edges; K=32 is exactly one bf16 WMMA step; A reused over 8 heads
__global__ __launch_bounds__(256)
void edge_score_kernel(const long long* __restrict__ src,
                       const long long* __restrict__ dst,
                       const __bf16* __restrict__ eabf,
                       const __bf16* __restrict__ wep,
                       const float* __restrict__ qb,
                       const float* __restrict__ kb,
                       float* __restrict__ score,
                       unsigned* __restrict__ smaxkey) {
  const int lane = threadIdx.x & 31;
  const int half = lane >> 4;
  const int lidx = lane & 15;
  // Wave-uniform wave index -> e0 uniform -> src/dst become scalar (SMEM) loads
  const int wv = __builtin_amdgcn_readfirstlane((int)(threadIdx.x >> 5));
  const int e0 = (blockIdx.x * 8 + wv) * 16;

  v16bf a = make_a_frag(eabf + (size_t)(e0 + lidx) * DED + half * 8);

  // Uniform index arrays (SGPRs), per-lane half-select into VGPRs
  int sO[16], dO[16];
#pragma unroll
  for (int j = 0; j < 16; ++j) {
    sO[j] = (int)src[e0 + j] * HO;   // pre-scaled row offsets (fit in 32 bits)
    dO[j] = (int)dst[e0 + j] * HO;
  }
  int sj[8], dj[8];
#pragma unroll
  for (int v = 0; v < 8; ++v) {
    sj[v] = half ? sO[v + 8] : sO[v];
    dj[v] = half ? dO[v + 8] : dO[v];
  }

  for (int h = 0; h < NH; ++h) {
    v16bf b = *(const v16bf*)(wep + ((size_t)h * 32 + lane) * 16);
    v8f e = {};
    e = __builtin_amdgcn_wmma_f32_16x16x32_bf16(false, a, false, b,
                                                (short)0, e, false, false);
    const int hl = h * 16 + lidx;
#pragma unroll
    for (int v = 0; v < 8; ++v) {
      // lane holds edge (v + 8*half), channel lidx of head h
      float kv = kb[sj[v] + hl];
      float qv = qb[dj[v] + hl];
      float t = qv * (kv + e[v]);
      // 16-lane reduction (xor of bits 0..3 stays inside the half-wave group)
      t += __shfl_xor(t, 1);
      t += __shfl_xor(t, 2);
      t += __shfl_xor(t, 4);
      t += __shfl_xor(t, 8);
      t *= 0.25f;  // 1/sqrt(OUT)
      if (lidx == 0) {
        int eg = e0 + v + 8 * half;
        score[eg * NH + h] = t;
        atomicMax(&smaxkey[(dj[v] >> 4) + h], fkey(t));  // dj*HO/16 = node*NH
      }
    }
  }
}

// ---------------- Kernel 4: exp(score - max) + segment-sum denominator ----
// one thread per edge: 1 dst load, vector score load/store, 8 exps + atomics
__global__ __launch_bounds__(256)
void softmax_norm_kernel(const long long* __restrict__ dst,
                         const unsigned* __restrict__ smaxkey,
                         float* __restrict__ score,
                         float* __restrict__ den) {
  int e = blockIdx.x * blockDim.x + threadIdx.x;
  if (e >= N_EDGES) return;
  int d = (int)dst[e] * NH;
  float4 s0 = ((const float4*)score)[e * 2];
  float4 s1 = ((const float4*)score)[e * 2 + 1];
  float v[8] = {s0.x, s0.y, s0.z, s0.w, s1.x, s1.y, s1.z, s1.w};
#pragma unroll
  for (int h = 0; h < NH; ++h) {
    float ex = __expf(v[h] - fkey_dec(smaxkey[d + h]));
    v[h] = ex;
    atomicAdd(&den[d + h], ex);
  }
  ((float4*)score)[e * 2]     = make_float4(v[0], v[1], v[2], v[3]);
  ((float4*)score)[e * 2 + 1] = make_float4(v[4], v[5], v[6], v[7]);
}

// ---------------- Kernel 5: edge pass 2 — recompute e (WMMA), scatter messages --
__global__ __launch_bounds__(256)
void edge_aggregate_kernel(const long long* __restrict__ src,
                           const long long* __restrict__ dst,
                           const __bf16* __restrict__ eabf,
                           const __bf16* __restrict__ wep,
                           const float* __restrict__ vb,
                           const float* __restrict__ ex,     // [E,NH] numerators
                           const float* __restrict__ den,    // [N,NH]
                           float* __restrict__ agg) {        // [N,HO]
  const int lane = threadIdx.x & 31;
  const int half = lane >> 4;
  const int lidx = lane & 15;
  const int wv = __builtin_amdgcn_readfirstlane((int)(threadIdx.x >> 5));
  const int e0 = (blockIdx.x * 8 + wv) * 16;

  v16bf a = make_a_frag(eabf + (size_t)(e0 + lidx) * DED + half * 8);

  int sO[16], dO[16];
#pragma unroll
  for (int j = 0; j < 16; ++j) {
    sO[j] = (int)src[e0 + j] * HO;
    dO[j] = (int)dst[e0 + j] * HO;
  }
  int sj[8], dj[8];
#pragma unroll
  for (int v = 0; v < 8; ++v) {
    sj[v] = half ? sO[v + 8] : sO[v];
    dj[v] = half ? dO[v + 8] : dO[v];
  }

  for (int h = 0; h < NH; ++h) {
    v16bf b = *(const v16bf*)(wep + ((size_t)h * 32 + lane) * 16);
    v8f e = {};
    e = __builtin_amdgcn_wmma_f32_16x16x32_bf16(false, a, false, b,
                                                (short)0, e, false, false);
    const int hl = h * 16 + lidx;
#pragma unroll
    for (int v = 0; v < 8; ++v) {
      int eg = e0 + v + 8 * half;
      float num = ex[eg * NH + h];
      float dn  = den[(dj[v] >> 4) + h];
      float alpha = num / fmaxf(dn, 1e-16f);
      float vv = vb[sj[v] + hl];
      atomicAdd(&agg[dj[v] + hl], (vv + e[v]) * alpha);
    }
  }
}

// ---------------- Kernel 6: head mean + skip + ReLU + residual ----------------
__global__ __launch_bounds__(256)
void finalize_kernel(const float* __restrict__ agg,
                     const float* __restrict__ s1,
                     const float* __restrict__ s2,
                     float* __restrict__ out) {
  long long i = (long long)blockIdx.x * blockDim.x + threadIdx.x;
  if (i >= (long long)N_NODES * DOUT) return;
  int n = (int)(i >> 4);
  int c = (int)(i & 15);
  float m = 0.0f;
#pragma unroll
  for (int h = 0; h < NH; ++h) m += agg[n * HO + h * 16 + c];
  m *= (1.0f / NH);
  float o = m + s1[i];
  out[i] = fmaxf(o, 0.0f) + s2[i];
}

// ---------------------------------------------------------------------------
extern "C" void kernel_launch(void* const* d_in, const int* in_sizes, int n_in,
                              void* d_out, int out_size, void* d_ws, size_t ws_size,
                              hipStream_t stream) {
  const float*     x     = (const float*)d_in[0];
  const long long* eidx  = (const long long*)d_in[1];   // int64 per reference
  const float*     eattr = (const float*)d_in[2];
  const float*     Wq    = (const float*)d_in[3];
  const float*     bq    = (const float*)d_in[4];
  const float*     Wk    = (const float*)d_in[5];
  const float*     bk    = (const float*)d_in[6];
  const float*     Wv    = (const float*)d_in[7];
  const float*     bv    = (const float*)d_in[8];
  const float*     We    = (const float*)d_in[9];
  const float*     Wskip = (const float*)d_in[10];
  const float*     bskip = (const float*)d_in[11];
  const float*     Wres  = (const float*)d_in[12];
  float*           out   = (float*)d_out;

  const long long* src = eidx;             // edge_index[0]
  const long long* dst = eidx + N_EDGES;   // edge_index[1]

  // Workspace carve (~205 MB total)
  char* w = (char*)d_ws;
  float*    qb      = (float*)w;    w += sizeof(float) * (size_t)N_NODES * HO;
  float*    kb      = (float*)w;    w += sizeof(float) * (size_t)N_NODES * HO;
  float*    vb      = (float*)w;    w += sizeof(float) * (size_t)N_NODES * HO;
  float*    agg     = (float*)w;    w += sizeof(float) * (size_t)N_NODES * HO;
  float*    s1      = (float*)w;    w += sizeof(float) * (size_t)N_NODES * DOUT;
  float*    s2      = (float*)w;    w += sizeof(float) * (size_t)N_NODES * DOUT;
  float*    den     = (float*)w;    w += sizeof(float) * (size_t)N_NODES * NH;
  unsigned* smaxkey = (unsigned*)w; w += sizeof(unsigned) * (size_t)N_NODES * NH;
  float*    score   = (float*)w;    w += sizeof(float) * (size_t)N_EDGES * NH;
  __bf16*   xbf     = (__bf16*)w;   w += sizeof(__bf16) * (size_t)N_NODES * DIN;
  __bf16*   eabf    = (__bf16*)w;   w += sizeof(__bf16) * (size_t)N_EDGES * DED;
  __bf16*   wqp     = (__bf16*)w;   w += sizeof(__bf16) * 8 * 4 * 32 * 16;
  __bf16*   wkp     = (__bf16*)w;   w += sizeof(__bf16) * 8 * 4 * 32 * 16;
  __bf16*   wvp     = (__bf16*)w;   w += sizeof(__bf16) * 8 * 4 * 32 * 16;
  __bf16*   wep     = (__bf16*)w;   w += sizeof(__bf16) * 8 * 1 * 32 * 16;
  __bf16*   wsp     = (__bf16*)w;   w += sizeof(__bf16) * 1 * 4 * 32 * 16;
  __bf16*   wrp     = (__bf16*)w;   w += sizeof(__bf16) * 1 * 4 * 32 * 16;

  // 0) precision/layout prep: bf16 activations + fragment-packed weights
  {
    long long nx = (long long)N_NODES * DIN / 2;
    cvt_bf16_kernel<<<(unsigned)((nx + 255) / 256), 256, 0, stream>>>(x, xbf, nx);
    long long ne = (long long)N_EDGES * DED / 2;
    cvt_bf16_kernel<<<(unsigned)((ne + 255) / 256), 256, 0, stream>>>(eattr, eabf, ne);
    pack_w_kernel<<<4, 256, 0, stream>>>(Wq,    wqp, DIN, 8, HO);
    pack_w_kernel<<<4, 256, 0, stream>>>(Wk,    wkp, DIN, 8, HO);
    pack_w_kernel<<<4, 256, 0, stream>>>(Wv,    wvp, DIN, 8, HO);
    pack_w_kernel<<<1, 256, 0, stream>>>(We,    wep, DED, 8, HO);
    pack_w_kernel<<<1, 256, 0, stream>>>(Wskip, wsp, DIN, 1, DOUT);
    pack_w_kernel<<<1, 256, 0, stream>>>(Wres,  wrp, DIN, 1, DOUT);
  }

  // 1) node projections: q,k,v + fused skip/res GEMMs (WMMA bf16)
  node_proj_kernel<<<dim3(N_NODES / 16, 26), 32, 0, stream>>>(
      xbf, wqp, wkp, wvp, wsp, wrp, bq, bk, bv, bskip, qb, kb, vb, s1, s2);

  // 2) init accumulators
  {
    long long n = (long long)N_NODES * HO;
    init_kernel<<<(unsigned)((n + 255) / 256), 256, 0, stream>>>(smaxkey, den, agg);
  }

  // 3) edge pass 1: e via WMMA, scores, segment max (50000 waves, 8/block)
  edge_score_kernel<<<N_EDGES / 16 / 8, 256, 0, stream>>>(
      src, dst, eabf, wep, qb, kb, score, smaxkey);

  // 4) softmax numerators + segment-sum denominators (one thread per edge)
  softmax_norm_kernel<<<(N_EDGES + 255) / 256, 256, 0, stream>>>(
      dst, smaxkey, score, den);

  // 5) edge pass 2: recompute e via WMMA, alpha-weighted scatter-add
  edge_aggregate_kernel<<<N_EDGES / 16 / 8, 256, 0, stream>>>(
      src, dst, eabf, wep, vb, score, den, agg);

  // 6) finalize: mean over heads + skip + ReLU + residual
  {
    long long n = (long long)N_NODES * DOUT;
    finalize_kernel<<<(unsigned)((n + 255) / 256), 256, 0, stream>>>(
        agg, s1, s2, out);
  }
}
